// MixtureOfExperts_37864431681938
// MI455X (gfx1250) — compile-verified, matrix operand order checked by reading, not run on hardware
//
#include <hip/hip_runtime.h>
#include <hip/hip_bf16.h>
#include <math.h>

// Problem constants (B=2, S=2048 -> T=4096)
#define TOK 4096
#define HD  1024
#define FD  4096
#define F2D 2048
#define NE  8

typedef __attribute__((ext_vector_type(16))) __bf16 v16bf;
typedef __attribute__((ext_vector_type(8)))  float  v8f;

__device__ __forceinline__ unsigned short f2bf(float f) {
  union { float f; unsigned int u; } v; v.f = f;
  unsigned int r = v.u + 0x7FFFu + ((v.u >> 16) & 1u);  // round-to-nearest-even
  return (unsigned short)(r >> 16);
}

__device__ __forceinline__ float gelu_exact(float x) {
  return 0.5f * x * (1.0f + erff(x * 0.70710678118654752440f));
}

// ---------------------------------------------------------------------------
// Fused LayerNorm + router (softmax -> top-2 -> renormalize)
// grid = TOK blocks, 256 threads
// ---------------------------------------------------------------------------
__global__ void moe_ln_router_kernel(const float* __restrict__ x,
                                     const float* __restrict__ g,
                                     const float* __restrict__ b,
                                     const float* __restrict__ rw,
                                     float* __restrict__ nx,
                                     float* __restrict__ wout) {
  const int t = blockIdx.x;
  const int tid = threadIdx.x;
  const float* xr = x + (size_t)t * HD;
  __shared__ float red[256];
  __shared__ float stat[2];
  __shared__ float lgs[NE];

  float s = 0.f;
  for (int i = tid; i < HD; i += 256) s += xr[i];
  red[tid] = s; __syncthreads();
  for (int o = 128; o > 0; o >>= 1) { if (tid < o) red[tid] += red[tid + o]; __syncthreads(); }
  if (tid == 0) stat[0] = red[0] * (1.0f / HD);
  __syncthreads();
  const float mean = stat[0];

  float v = 0.f;
  for (int i = tid; i < HD; i += 256) { float d = xr[i] - mean; v += d * d; }
  red[tid] = v; __syncthreads();
  for (int o = 128; o > 0; o >>= 1) { if (tid < o) red[tid] += red[tid + o]; __syncthreads(); }
  if (tid == 0) stat[1] = rsqrtf(red[0] * (1.0f / HD) + 1e-5f);
  __syncthreads();
  const float rstd = stat[1];

  float lg[NE];
#pragma unroll
  for (int e = 0; e < NE; ++e) lg[e] = 0.f;
  for (int i = tid; i < HD; i += 256) {
    float nv = (xr[i] - mean) * rstd * g[i] + b[i];
    nx[(size_t)t * HD + i] = nv;
#pragma unroll
    for (int e = 0; e < NE; ++e) lg[e] += nv * rw[i * NE + e];
  }
  for (int e = 0; e < NE; ++e) {
    red[tid] = lg[e]; __syncthreads();
    for (int o = 128; o > 0; o >>= 1) { if (tid < o) red[tid] += red[tid + o]; __syncthreads(); }
    if (tid == 0) lgs[e] = red[0];
    __syncthreads();
  }

  if (tid == 0) {
    float p[NE];
    float mx = -1e30f;
#pragma unroll
    for (int e = 0; e < NE; ++e) mx = fmaxf(mx, lgs[e]);
    float sm = 0.f;
#pragma unroll
    for (int e = 0; e < NE; ++e) { p[e] = expf(lgs[e] - mx); sm += p[e]; }
#pragma unroll
    for (int e = 0; e < NE; ++e) p[e] /= sm;
    int a1 = 0;
#pragma unroll
    for (int e = 1; e < NE; ++e) if (p[e] > p[a1]) a1 = e;
    int a2 = (a1 == 0) ? 1 : 0;
#pragma unroll
    for (int e = 0; e < NE; ++e) if (e != a1 && p[e] > p[a2]) a2 = e;
    const float dn = p[a1] + p[a2];
#pragma unroll
    for (int e = 0; e < NE; ++e)
      wout[(size_t)t * NE + e] = (e == a1 || e == a2) ? (p[e] / dn) : 0.f;
  }
}

// ---------------------------------------------------------------------------
// Row-wise LayerNorm (optionally followed by exact GELU), in place.
// grid = TOK blocks, 256 threads
// ---------------------------------------------------------------------------
__global__ void moe_ln_act_rows_kernel(float* __restrict__ z,
                                       const float* __restrict__ g,
                                       const float* __restrict__ b,
                                       int N, int doGelu) {
  const int t = blockIdx.x;
  const int tid = threadIdx.x;
  float* zr = z + (size_t)t * N;
  __shared__ float red[256];
  __shared__ float stat[2];

  float s = 0.f;
  for (int i = tid; i < N; i += 256) s += zr[i];
  red[tid] = s; __syncthreads();
  for (int o = 128; o > 0; o >>= 1) { if (tid < o) red[tid] += red[tid + o]; __syncthreads(); }
  if (tid == 0) stat[0] = red[0] / (float)N;
  __syncthreads();
  const float mean = stat[0];

  float v = 0.f;
  for (int i = tid; i < N; i += 256) { float d = zr[i] - mean; v += d * d; }
  red[tid] = v; __syncthreads();
  for (int o = 128; o > 0; o >>= 1) { if (tid < o) red[tid] += red[tid + o]; __syncthreads(); }
  if (tid == 0) stat[1] = rsqrtf(red[0] / (float)N + 1e-5f);
  __syncthreads();
  const float rstd = stat[1];

  for (int i = tid; i < N; i += 256) {
    float val = (zr[i] - mean) * rstd * g[i] + b[i];
    zr[i] = doGelu ? gelu_exact(val) : val;
  }
}

// ---------------------------------------------------------------------------
// y = x (residual init)
// ---------------------------------------------------------------------------
__global__ void moe_init_out_kernel(const float* __restrict__ x,
                                    float* __restrict__ y, int n) {
  int i = blockIdx.x * blockDim.x + threadIdx.x;
  if (i < n) y[i] = x[i];
}

// ---------------------------------------------------------------------------
// Generic bf16 WMMA GEMM:  C[M,N] = epi(A[M,K] @ B[K,N] + bias[N])
// Tiles: 128x128x32; 256 threads = 8 waves, wave grid 2(M) x 4(N),
// each wave 64x32 = 4x2 fragments of v_wmma_f32_16x16x32_bf16.
// Double-buffered LDS (ping-pong) + global_prefetch 2 slices ahead.
// mode 0: store acc+bias          (C)
// mode 1: store gelu(acc+bias)    (C)
// mode 2: y[m,n] += rwt[m,eidx] * (acc+bias)   (combine, C unused)
// Requires M%128==0, N%128==0, K%32==0 (true for all launches here).
// ---------------------------------------------------------------------------
__global__ __launch_bounds__(256) void moe_wmma_gemm_kernel(
    const float* __restrict__ A, const float* __restrict__ Bw,
    const float* __restrict__ bias, float* __restrict__ C,
    int M, int N, int Kd, int mode,
    const float* __restrict__ rwt, int eidx, float* __restrict__ y) {
  constexpr int BM = 128, BN = 128, BK = 32;
  __shared__ alignas(32) unsigned short As[2][BM * BK];   // [buf][row][k]
  __shared__ alignas(32) unsigned short Bst[2][BN * BK];  // [buf][n][k] (transposed)

  const int tid = threadIdx.x;
  const int mBase = blockIdx.y * BM;
  const int nBase = blockIdx.x * BN;
  const int lane = tid & 31;
  const int wv = tid >> 5;
  const int wm = (wv & 1) * 64;    // wave M offset in block tile
  const int wn = (wv >> 1) * 32;   // wave N offset in block tile
  const int lr = lane & 15;        // row/col within fragment
  const int lh = (lane >> 4) * 16; // K half: lanes 0-15 -> K 0-15, 16-31 -> K 16-31

  // Per-thread staging coordinates (each thread moves 4x float4)
  const int ar = tid >> 1;               // A: 128 rows, 2 threads/row
  const int ac = (tid & 1) * 16;         // A: two 16-col halves
  const int bk = tid & 31;               // B: 32 K rows
  const int bc = (tid >> 5) * 16;        // B: eight 16-col chunks

  auto loadA = [&](int kb, float4* r) {
#pragma unroll
    for (int i = 0; i < 4; ++i)
      r[i] = *(const float4*)(A + (size_t)(mBase + ar) * Kd + kb + ac + i * 4);
  };
  auto loadB = [&](int kb, float4* r) {
#pragma unroll
    for (int i = 0; i < 4; ++i)
      r[i] = *(const float4*)(Bw + (size_t)(kb + bk) * N + nBase + bc + i * 4);
  };
  auto storeA = [&](int buf, const float4* r) {
#pragma unroll
    for (int i = 0; i < 4; ++i) {
      As[buf][ar * BK + ac + i * 4 + 0] = f2bf(r[i].x);
      As[buf][ar * BK + ac + i * 4 + 1] = f2bf(r[i].y);
      As[buf][ar * BK + ac + i * 4 + 2] = f2bf(r[i].z);
      As[buf][ar * BK + ac + i * 4 + 3] = f2bf(r[i].w);
    }
  };
  auto storeB = [&](int buf, const float4* r) {
#pragma unroll
    for (int i = 0; i < 4; ++i) {
      Bst[buf][(bc + i * 4 + 0) * BK + bk] = f2bf(r[i].x);
      Bst[buf][(bc + i * 4 + 1) * BK + bk] = f2bf(r[i].y);
      Bst[buf][(bc + i * 4 + 2) * BK + bk] = f2bf(r[i].z);
      Bst[buf][(bc + i * 4 + 3) * BK + bk] = f2bf(r[i].w);
    }
  };

  v8f zero = {};
  v8f acc[4][2];
#pragma unroll
  for (int mi = 0; mi < 4; ++mi)
#pragma unroll
    for (int ni = 0; ni < 2; ++ni) acc[mi][ni] = zero;

  // Prologue: stage slice 0 into buffer 0
  {
    float4 a0[4], b0[4];
    loadA(0, a0);
    loadB(0, b0);
    storeA(0, a0);
    storeB(0, b0);
  }
  __syncthreads();

  for (int kb = 0; kb < Kd; kb += BK) {
    const int cur = (kb / BK) & 1;
    const int nxt = cur ^ 1;
    const bool more = (kb + BK) < Kd;

    // Warm GL2 two slices ahead (global_prefetch_b8 path)
    if (kb + 2 * BK < Kd) {
      __builtin_prefetch(A + (size_t)(mBase + ar) * Kd + kb + 2 * BK + ac, 0, 3);
      __builtin_prefetch(Bw + (size_t)(kb + 2 * BK + bk) * N + nBase + bc, 0, 3);
    }

    // Issue next slice's global loads (latency hidden behind WMMAs)
    float4 aNext[4], bNext[4];
    if (more) {
      loadA(kb + BK, aNext);
      loadB(kb + BK, bNext);
    }

    // Compute on current buffer
    v16bf aF[4], bF[2];
#pragma unroll
    for (int mi = 0; mi < 4; ++mi)
      aF[mi] = *reinterpret_cast<const v16bf*>(&As[cur][(wm + mi * 16 + lr) * BK + lh]);
#pragma unroll
    for (int ni = 0; ni < 2; ++ni)
      bF[ni] = *reinterpret_cast<const v16bf*>(&Bst[cur][(wn + ni * 16 + lr) * BK + lh]);

#pragma unroll
    for (int mi = 0; mi < 4; ++mi)
#pragma unroll
      for (int ni = 0; ni < 2; ++ni)
        acc[mi][ni] = __builtin_amdgcn_wmma_f32_16x16x32_bf16(
            false, aF[mi], false, bF[ni], (short)0, acc[mi][ni], false, false);

    // Stage next slice into the other buffer (disjoint from 'cur' being read)
    if (more) {
      storeA(nxt, aNext);
      storeB(nxt, bNext);
    }
    __syncthreads();
  }

  // Epilogue. C/D layout: VGPR r -> M = r + 8*(lane>=16), N = lane%16
  const int mrb = (lane >> 4) * 8;
  const int nc = lane & 15;
#pragma unroll
  for (int mi = 0; mi < 4; ++mi) {
#pragma unroll
    for (int ni = 0; ni < 2; ++ni) {
#pragma unroll
      for (int r = 0; r < 8; ++r) {
        int m = mBase + wm + mi * 16 + mrb + r;
        int n = nBase + wn + ni * 16 + nc;
        float v = acc[mi][ni][r] + bias[n];
        if (mode == 1) v = gelu_exact(v);
        if (mode == 2) {
          y[(size_t)m * N + n] += rwt[(size_t)m * NE + eidx] * v;
        } else {
          C[(size_t)m * N + n] = v;
        }
      }
    }
  }
}

// ---------------------------------------------------------------------------
// Host orchestration
// ---------------------------------------------------------------------------
extern "C" void kernel_launch(void* const* d_in, const int* in_sizes, int n_in,
                              void* d_out, int out_size, void* d_ws, size_t ws_size,
                              hipStream_t stream) {
  (void)in_sizes; (void)n_in; (void)out_size; (void)ws_size;

  const float* x        = (const float*)d_in[0];
  const float* ln_g     = (const float*)d_in[1];
  const float* ln_b     = (const float*)d_in[2];
  const float* router_w = (const float*)d_in[3];
  const float* up_W     = (const float*)d_in[4];
  const float* up_b     = (const float*)d_in[5];
  const float* down_W   = (const float*)d_in[6];
  const float* down_b   = (const float*)d_in[7];
  const float* spec0_W  = (const float*)d_in[8];
  const float* spec0_b  = (const float*)d_in[9];
  const float* ln0_g    = (const float*)d_in[10];
  const float* ln0_b    = (const float*)d_in[11];
  const float* spec1a_W = (const float*)d_in[12];
  const float* spec1a_b = (const float*)d_in[13];
  const float* spec1b_W = (const float*)d_in[14];
  const float* spec1b_b = (const float*)d_in[15];
  const float* ln1_g    = (const float*)d_in[16];
  const float* ln1_b    = (const float*)d_in[17];
  const float* spec2_W  = (const float*)d_in[18];
  const float* spec2_b  = (const float*)d_in[19];

  float* y  = (float*)d_out;
  float* ws = (float*)d_ws;

  // Workspace layout (floats): nx[T,H], w[T,E], h1[T,F], z[T,F], t[T,F/2]
  float* nx  = ws;                size_t off = (size_t)TOK * HD;
  float* wrt = ws + off;          off += (size_t)TOK * NE;
  float* h1  = ws + off;          off += (size_t)TOK * FD;
  float* zb  = ws + off;          off += (size_t)TOK * FD;
  float* tb  = ws + off;          off += (size_t)TOK * F2D;

  moe_ln_router_kernel<<<TOK, 256, 0, stream>>>(x, ln_g, ln_b, router_w, nx, wrt);
  moe_init_out_kernel<<<(TOK * HD) / 256, 256, 0, stream>>>(x, y, TOK * HD);

  auto gemm = [&](const float* A, const float* B, const float* bias, float* C,
                  int M, int N, int K, int mode, int eidx) {
    dim3 grid(N / 128, M / 128);
    moe_wmma_gemm_kernel<<<grid, 256, 0, stream>>>(A, B, bias, C, M, N, K,
                                                   mode, wrt, eidx, y);
  };

  for (int e = 0; e < NE; ++e) {
    const int j = e / 4;      // which of the 2 spec weight rows
    const int ty = e % 4;     // specialization type

    // h1 = gelu(nx @ up_W[e] + up_b[e])
    gemm(nx, up_W + (size_t)e * HD * FD, up_b + (size_t)e * FD, h1,
         TOK, FD, HD, /*gelu*/ 1, e);

    const float* src = h1;
    if (ty == 0) {
      // s0 = gelu(ln(h1 @ spec0_W[j] + b))
      gemm(h1, spec0_W + (size_t)j * FD * FD, spec0_b + (size_t)j * FD, zb,
           TOK, FD, FD, 0, e);
      moe_ln_act_rows_kernel<<<TOK, 256, 0, stream>>>(
          zb, ln0_g + (size_t)j * FD, ln0_b + (size_t)j * FD, FD, /*gelu*/ 1);
      src = zb;
    } else if (ty == 1) {
      // s1 = ln( gelu(h1 @ spec1a[j] + b) @ spec1b[j] + b )
      gemm(h1, spec1a_W + (size_t)j * FD * F2D, spec1a_b + (size_t)j * F2D, tb,
           TOK, F2D, FD, /*gelu*/ 1, e);
      gemm(tb, spec1b_W + (size_t)j * F2D * FD, spec1b_b + (size_t)j * FD, zb,
           TOK, FD, F2D, 0, e);
      moe_ln_act_rows_kernel<<<TOK, 256, 0, stream>>>(
          zb, ln1_g + (size_t)j * FD, ln1_b + (size_t)j * FD, FD, /*gelu*/ 0);
      src = zb;
    } else if (ty == 2) {
      // s2 = gelu(h1 @ spec2[j] + b)
      gemm(h1, spec2_W + (size_t)j * FD * FD, spec2_b + (size_t)j * FD, zb,
           TOK, FD, FD, /*gelu*/ 1, e);
      src = zb;
    } // ty == 3: identity, src stays h1

    // y += w[:,e] * (src @ down_W[e] + down_b[e])
    gemm(src, down_W + (size_t)e * FD * HD, down_b + (size_t)e * HD, nullptr,
         TOK, HD, FD, /*combine*/ 2, e);
  }
}
